// SelfAttentionLayer_11287174054463
// MI455X (gfx1250) — matrix operand order repaired and checked by reading, not compiled, not run on hardware
//
#include <hip/hip_runtime.h>

// ---------------------------------------------------------------------------
// Problem constants (reference: B=16, N=2048, C=512)
// ---------------------------------------------------------------------------
#define BQ 16
#define NQ 2048
#define CQ 512
#define MROWS (BQ * NQ)          // 32768 total rows
#define EPS_NUM 1e-9f
#define BN_EPS 1e-5f

typedef __attribute__((ext_vector_type(16))) _Float16 v16h;
typedef __attribute__((ext_vector_type(8)))  _Float16 v8h;
typedef __attribute__((ext_vector_type(8)))  float    v8f;

#if __has_builtin(__builtin_amdgcn_global_load_async_to_lds_b128)
#define HAVE_ASYNC_LDS 1
typedef int v4i __attribute__((ext_vector_type(4)));
typedef __attribute__((address_space(1))) v4i  gv4i;   // global int4
typedef __attribute__((address_space(3))) v4i  lv4i;   // LDS int4
typedef __attribute__((address_space(1))) void gvoid;
typedef __attribute__((address_space(3))) void lvoid;
#endif

// ---------------------------------------------------------------------------
// WMMA fragment loaders (CDNA5 wave32 layouts, CDNA5 ISA §7.12.2)
//
// A (16x32 f16, row major, leading dim ld):
//   lanes 0-15  : row=lane,     VGPR0-3 -> K=0..7,  VGPR4-7 -> K=16..23
//   lanes 16-31 : row=lane-16,  VGPR0-3 -> K=8..15, VGPR4-7 -> K=24..31
//   => per lane: two contiguous 16B chunks at K = h*8 and K = 16 + h*8
// ---------------------------------------------------------------------------
__device__ __forceinline__ v16h frag_a_rowmajor(const _Float16* tile, int ld, int lane) {
  const int r = lane & 15, h = (lane >> 4) & 1;
  const _Float16* p = tile + (size_t)r * ld + h * 8;
  v8h lo = *(const v8h*)(p);
  v8h hi = *(const v8h*)(p + 16);
  v16h a;
#pragma unroll
  for (int i = 0; i < 8; ++i) { a[i] = lo[i]; a[i + 8] = hi[i]; }
  return a;
}

// B (32x16 f16, K x N). Source is the *transposed* operand T[n][k] row major
// (weights pre-transposed to [cout][cin]; k-matrix is naturally [n][c]).
//   lanes 0-15  : col=lane,    VGPR0-3 -> K=0..7,  VGPR4-7 -> K=16..23
//   lanes 16-31 : col=lane-16, VGPR0-3 -> K=8..15, VGPR4-7 -> K=24..31
//   => per lane: two contiguous 16B chunks at K = h*8 and K = 16 + h*8
__device__ __forceinline__ v16h frag_b_from_T(const _Float16* tileT, int ld, int lane) {
  const int n = lane & 15, h = (lane >> 4) & 1;
  const _Float16* p = tileT + (size_t)n * ld + h * 8;
  v8h lo = *(const v8h*)(p);
  v8h hi = *(const v8h*)(p + 16);
  v16h b;
#pragma unroll
  for (int i = 0; i < 8; ++i) { b[i] = lo[i]; b[i + 8] = hi[i]; }
  return b;
}

// ---------------------------------------------------------------------------
// Prep kernels: f32 -> f16 conversion, weight transpose, col zeroing
// ---------------------------------------------------------------------------
__global__ __launch_bounds__(256) void convert_x_kernel(const float* __restrict__ x,
                                                        _Float16* __restrict__ xh) {
  size_t i = (size_t)blockIdx.x * 256 + threadIdx.x;   // exactly MROWS*CQ threads
  xh[i] = (_Float16)x[i];
}

__global__ __launch_bounds__(256) void transpose_w_kernel(const float* __restrict__ W,
                                                          _Float16* __restrict__ Wt) {
  int i = blockIdx.x * 256 + threadIdx.x;              // exactly 512*512 threads
  int co = i >> 9, ci = i & 511;
  Wt[(size_t)co * CQ + ci] = (_Float16)W[(size_t)ci * CQ + co];
}

__global__ __launch_bounds__(256) void zero_col_kernel(float* __restrict__ col) {
  int i = blockIdx.x * 256 + threadIdx.x;              // exactly B*N threads
  col[i] = 0.f;
}

// ---------------------------------------------------------------------------
// Projection GEMM: O[M=32768, 512] = A[M,512](f16) @ Wt[cout][cin](f16)^T + bias
// One wave per 16x16 tile, K-loop of 16 x wmma_f32_16x16x32_f16.
// store_f32==0 -> f16 output (q,k,v); store_f32==1 -> f32 output (z = v@Wo).
// ---------------------------------------------------------------------------
__global__ __launch_bounds__(256) void gemm512_kernel(
    const _Float16* __restrict__ A, const _Float16* __restrict__ Wt,
    const float* __restrict__ bias, _Float16* __restrict__ Oh,
    float* __restrict__ Of, int store_f32)
{
  const int lane = threadIdx.x & 31;
  const int wid  = blockIdx.x * 8 + (threadIdx.x >> 5);
  const int tm   = wid >> 5;   // 0..2047  (M tiles)
  const int tn   = wid & 31;   // 0..31    (Cout tiles)
  const _Float16* Abase = A  + (size_t)tm * 16 * CQ;
  const _Float16* Bbase = Wt + (size_t)tn * 16 * CQ;

  v8f acc = {};
#pragma unroll
  for (int kk = 0; kk < CQ; kk += 32) {
    v16h a = frag_a_rowmajor(Abase + kk, CQ, lane);
    v16h b = frag_b_from_T(Bbase + kk, CQ, lane);
    acc = __builtin_amdgcn_wmma_f32_16x16x32_f16(false, a, false, b,
                                                 (short)0, acc, false, false);
  }

  const int n  = lane & 15;
  const int mo = (lane >> 4) * 8;
  const float bv = bias ? bias[tn * 16 + n] : 0.f;
#pragma unroll
  for (int v = 0; v < 8; ++v) {
    size_t idx = (size_t)(tm * 16 + mo + v) * CQ + (size_t)tn * 16 + n;
    float val = acc[v] + bv;
    if (store_f32) Of[idx] = val;
    else           Oh[idx] = (_Float16)val;
  }
}

// ---------------------------------------------------------------------------
// Attention column sums, register-blocked:
//   grid (N/64, B); block 128 = 4 waves. The 64-row q block lives in LDS.
//   Each wave computes 64x64 score tiles (4x4 wmma accumulators) for every
//   4th 64-column group: 16 wmma per 16 b128 loads (vs 1:4 unblocked).
//   Pass 0: per-wave online row max/sumexp partials -> merge across the
//           16-lane half (ds-bpermute shuffles) -> merge across waves in LDS.
//   Pass 1: recompute scores, p = exp(s-max)/Z; columns partition exactly
//           across (wave, group, lane) so col_s needs plain stores only,
//           then one coalesced atomic flush to global col[b][n].
// ---------------------------------------------------------------------------
#define ATT_BM 64
#define LDQ 520   // 512 + 8 halves pad -> consecutive rows rotate LDS banks

__device__ __forceinline__ void score_tile64(
    const _Float16* qtile,          // LDS, 64 rows x LDQ
    const _Float16* kbase,          // kh rows g*64.. (row major [n][c])
    const _Float16* knext,          // next group's rows (prefetch hint) or null
    int lane, v8f (&acc)[4][4])
{
  const v8f zero = {};
#pragma unroll
  for (int rt = 0; rt < 4; ++rt)
#pragma unroll
    for (int ct = 0; ct < 4; ++ct) acc[rt][ct] = zero;

  if (knext) __builtin_prefetch(knext + (size_t)lane * 1024, 0, 1);

  for (int kk = 0; kk < CQ; kk += 32) {
    v16h a[4], bf[4];
#pragma unroll
    for (int rt = 0; rt < 4; ++rt)
      a[rt] = frag_a_rowmajor(qtile + rt * 16 * LDQ + kk, LDQ, lane);
#pragma unroll
    for (int ct = 0; ct < 4; ++ct)
      bf[ct] = frag_b_from_T(kbase + (size_t)ct * 16 * CQ + kk, CQ, lane);
#pragma unroll
    for (int rt = 0; rt < 4; ++rt)
#pragma unroll
      for (int ct = 0; ct < 4; ++ct)
        acc[rt][ct] = __builtin_amdgcn_wmma_f32_16x16x32_f16(
            false, a[rt], false, bf[ct], (short)0, acc[rt][ct], false, false);
  }
}

__global__ __launch_bounds__(128) void attn_colsum_kernel(
    const _Float16* __restrict__ qh, const _Float16* __restrict__ kh,
    float* __restrict__ col)
{
  __shared__ __align__(16) _Float16 q_s[ATT_BM * LDQ];
  __shared__ float col_s[NQ];
  __shared__ float stats_s[4][ATT_BM][2];   // per-wave {rowmax, rowsum}

  const int tid  = threadIdx.x;
  const int lane = tid & 31;
  const int w    = tid >> 5;                 // wave 0..3
  const int mo   = (lane >> 4) * 8;
  const int b    = blockIdx.y;
  const int mblk = blockIdx.x;
  const size_t qrow0 = (size_t)b * NQ + (size_t)mblk * ATT_BM;

  // ---- stage q block (64 x 512 f16) into padded LDS --------------------
  {
    const _Float16* src = qh + qrow0 * CQ;
#if defined(HAVE_ASYNC_LDS)
    for (int ch = tid; ch < ATT_BM * (CQ / 8); ch += 128) {
      int row = ch >> 6, kc = ch & 63;
      __builtin_amdgcn_global_load_async_to_lds_b128(
          (gv4i*)(gvoid*)(void*)(const_cast<_Float16*>(src + row * CQ + kc * 8)),
          (lv4i*)(lvoid*)(q_s + row * LDQ + kc * 8), 0, 0);
    }
#if __has_builtin(__builtin_amdgcn_s_wait_asynccnt)
    __builtin_amdgcn_s_wait_asynccnt(0);
#else
    asm volatile("s_wait_asynccnt 0x0" ::: "memory");
#endif
#else
    for (int ch = tid; ch < ATT_BM * (CQ / 8); ch += 128) {
      int row = ch >> 6, kc = ch & 63;
      *(v8h*)(q_s + row * LDQ + kc * 8) = *(const v8h*)(src + row * CQ + kc * 8);
    }
#endif
  }
  __syncthreads();

  const _Float16* kb0 = kh + (size_t)b * NQ * CQ;

  float rmax[4][8], rsum[4][8];
#pragma unroll
  for (int rt = 0; rt < 4; ++rt)
#pragma unroll
    for (int v = 0; v < 8; ++v) { rmax[rt][v] = -3.0e38f; rsum[rt][v] = 0.f; }

  // ---- pass 0: per-wave online softmax stats over this wave's groups ---
  for (int j = 0; j < 8; ++j) {
    const int g = w + 4 * j;                     // 64-col group index, 0..31
    v8f acc[4][4];
    const _Float16* knext = (j + 1 < 8) ? kb0 + (size_t)(g + 4) * 64 * CQ : nullptr;
    score_tile64(q_s, kb0 + (size_t)g * 64 * CQ, knext, lane, acc);
#pragma unroll
    for (int rt = 0; rt < 4; ++rt)
#pragma unroll
      for (int v = 0; v < 8; ++v) {
        float a0 = acc[rt][0][v], a1 = acc[rt][1][v];
        float a2 = acc[rt][2][v], a3 = acc[rt][3][v];
        float m4 = fmaxf(fmaxf(a0, a1), fmaxf(a2, a3));
        float s4 = __expf(a0 - m4) + __expf(a1 - m4) +
                   __expf(a2 - m4) + __expf(a3 - m4);
        float nm = fmaxf(rmax[rt][v], m4);
        rsum[rt][v] = rsum[rt][v] * __expf(rmax[rt][v] - nm) +
                      s4 * __expf(m4 - nm);
        rmax[rt][v] = nm;
      }
  }

  // merge per-column-slot partials across the 16-lane half
#pragma unroll
  for (int mask = 1; mask <= 8; mask <<= 1) {
#pragma unroll
    for (int rt = 0; rt < 4; ++rt)
#pragma unroll
      for (int v = 0; v < 8; ++v) {
        float om = __shfl_xor(rmax[rt][v], mask, 32);
        float os = __shfl_xor(rsum[rt][v], mask, 32);
        float nm = fmaxf(rmax[rt][v], om);
        rsum[rt][v] = rsum[rt][v] * __expf(rmax[rt][v] - nm) +
                      os * __expf(om - nm);
        rmax[rt][v] = nm;
      }
  }
  // publish per-wave stats (lanes 0 and 16 hold rows mo..mo+7 of each rt)
  if ((lane & 15) == 0) {
#pragma unroll
    for (int rt = 0; rt < 4; ++rt)
#pragma unroll
      for (int v = 0; v < 8; ++v) {
        stats_s[w][rt * 16 + mo + v][0] = rmax[rt][v];
        stats_s[w][rt * 16 + mo + v][1] = rsum[rt][v];
      }
  }
  __syncthreads();

  // merge the 4 waves' partials -> final rowmax and 1/Z
#pragma unroll
  for (int rt = 0; rt < 4; ++rt)
#pragma unroll
    for (int v = 0; v < 8; ++v) {
      const int row = rt * 16 + mo + v;
      float m = -3.0e38f, s = 0.f;
#pragma unroll
      for (int w2 = 0; w2 < 4; ++w2) {
        float om = stats_s[w2][row][0];
        float os = stats_s[w2][row][1];
        float nm = fmaxf(m, om);
        s = s * __expf(m - nm) + os * __expf(om - nm);
        m = nm;
      }
      rmax[rt][v] = m;
      rsum[rt][v] = 1.f / s;                  // Z >= 1 always
    }

  // ---- pass 1: recompute scores, accumulate column sums ----------------
  for (int j = 0; j < 8; ++j) {
    const int g = w + 4 * j;
    v8f acc[4][4];
    const _Float16* knext = (j + 1 < 8) ? kb0 + (size_t)(g + 4) * 64 * CQ : nullptr;
    score_tile64(q_s, kb0 + (size_t)g * 64 * CQ, knext, lane, acc);
#pragma unroll
    for (int ct = 0; ct < 4; ++ct) {
      float cp = 0.f;
#pragma unroll
      for (int rt = 0; rt < 4; ++rt)
#pragma unroll
        for (int v = 0; v < 8; ++v)
          cp += __expf(acc[rt][ct][v] - rmax[rt][v]) * rsum[rt][v];
      cp += __shfl_xor(cp, 16, 32);           // rows 0-7 half + rows 8-15 half
      if (lane < 16)                          // each column written exactly once
        col_s[g * 64 + ct * 16 + lane] = cp;
    }
  }

  __syncthreads();
  for (int i = tid; i < NQ; i += 128)
    atomicAdd(&col[(size_t)b * NQ + i], col_s[i]);
}

// ---------------------------------------------------------------------------
// Fused epilogue: scale = col/(eps+col); y = scale*z + bo; BN; ReLU; +x
// ---------------------------------------------------------------------------
__global__ __launch_bounds__(256) void finalize_kernel(
    const float* __restrict__ z, const float* __restrict__ col,
    const float* __restrict__ x, const float* __restrict__ bo,
    const float* __restrict__ gamma, const float* __restrict__ beta,
    const float* __restrict__ mean, const float* __restrict__ var,
    float* __restrict__ out)
{
  size_t i = (size_t)blockIdx.x * 256 + threadIdx.x;   // exactly MROWS*CQ
  int c = (int)(i & (CQ - 1));
  size_t r = i >> 9;
  float cv = col[r];
  float s  = cv / (EPS_NUM + cv);
  float y  = s * z[i] + bo[c];
  y = (y - mean[c]) * rsqrtf(var[c] + BN_EPS) * gamma[c] + beta[c];
  out[i] = fmaxf(y, 0.f) + x[i];
}

// ---------------------------------------------------------------------------
// Host launcher
// ---------------------------------------------------------------------------
extern "C" void kernel_launch(void* const* d_in, const int* in_sizes, int n_in,
                              void* d_out, int out_size, void* d_ws, size_t ws_size,
                              hipStream_t stream) {
  (void)in_sizes; (void)n_in; (void)out_size; (void)ws_size;

  const float* x    = (const float*)d_in[0];
  const float* Wq   = (const float*)d_in[1];
  const float* bq   = (const float*)d_in[2];
  const float* Wk   = (const float*)d_in[3];
  const float* bk   = (const float*)d_in[4];
  const float* Wv   = (const float*)d_in[5];
  const float* bv   = (const float*)d_in[6];
  const float* Wo   = (const float*)d_in[7];
  const float* bo   = (const float*)d_in[8];
  const float* g    = (const float*)d_in[9];
  const float* be   = (const float*)d_in[10];
  const float* mean = (const float*)d_in[11];
  const float* var  = (const float*)d_in[12];
  float* out = (float*)d_out;

  // workspace carve-up (~130 MB). zf (f32, 67MB) aliases xh+qh, which are dead
  // by the time z = v@Wo runs (attention consumes qh/kh before that GEMM).
  char* ws = (char*)d_ws;
  const size_t SZ_H = (size_t)MROWS * CQ * sizeof(_Float16);   // 33,554,432 B
  _Float16* xh = (_Float16*)(ws);
  _Float16* qh = (_Float16*)(ws + SZ_H);
  _Float16* kh = (_Float16*)(ws + 2 * SZ_H);
  _Float16* vh = (_Float16*)(ws + 3 * SZ_H);
  float*    zf = (float*)(ws);                                  // alias xh..qh
  _Float16* wt = (_Float16*)(ws + 4 * SZ_H);
  const size_t WSZ = (size_t)CQ * CQ;                           // elems / weight
  _Float16 *wtq = wt, *wtk = wt + WSZ, *wtv = wt + 2 * WSZ, *wto = wt + 3 * WSZ;
  float* col = (float*)(ws + 4 * SZ_H + 4 * WSZ * sizeof(_Float16));

  // 1) precision staging
  convert_x_kernel<<<MROWS * CQ / 256, 256, 0, stream>>>(x, xh);
  transpose_w_kernel<<<WSZ / 256, 256, 0, stream>>>(Wq, wtq);
  transpose_w_kernel<<<WSZ / 256, 256, 0, stream>>>(Wk, wtk);
  transpose_w_kernel<<<WSZ / 256, 256, 0, stream>>>(Wv, wtv);
  transpose_w_kernel<<<WSZ / 256, 256, 0, stream>>>(Wo, wto);

  // 2) projections (WMMA f16 -> f32 acc -> f16)
  const int GEMM_BLOCKS = (MROWS / 16) * (CQ / 16) / 8;         // 8192
  gemm512_kernel<<<GEMM_BLOCKS, 256, 0, stream>>>(xh, wtq, bq, qh, nullptr, 0);
  gemm512_kernel<<<GEMM_BLOCKS, 256, 0, stream>>>(xh, wtk, bk, kh, nullptr, 0);
  gemm512_kernel<<<GEMM_BLOCKS, 256, 0, stream>>>(xh, wtv, bv, vh, nullptr, 0);

  // 3) softmax column sums (69 GFLOP per score pass), then z = v@Wo into aliased ws
  zero_col_kernel<<<BQ * NQ / 256, 256, 0, stream>>>(col);
  attn_colsum_kernel<<<dim3(NQ / ATT_BM, BQ), 128, 0, stream>>>(qh, kh, col);
  gemm512_kernel<<<GEMM_BLOCKS, 256, 0, stream>>>(vh, wto, nullptr, nullptr, zf, 1);

  // 4) fused scale + bias + BatchNorm + ReLU + residual
  finalize_kernel<<<MROWS * CQ / 256, 256, 0, stream>>>(zf, col, x, bo, g, be,
                                                        mean, var, out);
}
